// LinearAlt_15135464751426
// MI455X (gfx1250) — compile-verified, matrix operand order checked by reading, not run on hardware
//
#include <hip/hip_runtime.h>

typedef __attribute__((ext_vector_type(16))) _Float16 v16h;
typedef __attribute__((ext_vector_type(8)))  float    v8f;

// quant_w(v) = round(v*128)/128; rintf = round-half-to-even = jnp.round
__device__ __forceinline__ float dq(float v) {
    return rintf(v * 128.0f) * 0.0078125f;
}

// Native CDNA5 V_TANH_F32 (TRANS op) instead of branchy ocml tanhf.
__device__ __forceinline__ float fast_tanh(float x) {
#if __has_builtin(__builtin_amdgcn_tanhf)
    return __builtin_amdgcn_tanhf(x);
#elif __has_builtin(__builtin_amdgcn_tanh_f32)
    return __builtin_amdgcn_tanh_f32(x);
#else
    float r;
    asm volatile("v_tanh_f32_e32 %0, %1" : "=v"(r) : "v"(x));
    return r;
#endif
}

// ---------------------------------------------------------------------------
// Kernel 1: W'[o,k] = (f16)( quant(weight[o,k]) - quant(mask[o,k]) ), 256x256.
// Exact multiples of 1/128, |v| <~ 0.2 -> exactly representable in f16.
// ---------------------------------------------------------------------------
__global__ void build_w_f16(const float* __restrict__ weight,
                            const float* __restrict__ mask,
                            _Float16* __restrict__ Wh) {
    const int i = blockIdx.x * blockDim.x + threadIdx.x;    // 0 .. 65535
    Wh[i] = (_Float16)(dq(weight[i]) - dq(mask[i]));
}

// ---------------------------------------------------------------------------
// Kernel 2: out = tanh( (x - quant(mask)) @ W'^T ) + quant(mask)
// M = 131072, N = 256, K = 256.  256 threads = 8 waves per block.
//
// Per block: stage the whole 256x256 f16 W' into LDS once (132 KB with
// 264-half row padding -> conflict-free ds_load_b128: stride 528 B, bank
// start 4*lane mod 64). Then each wave computes a 16(M) x 256(N) strip:
// 8 register-resident A fragments, 16 N-tiles x 8 v_wmma_f32_16x16x32_f16
// fed from LDS, epilogue = native v_tanh_f32 + quantized out_mask.
// ---------------------------------------------------------------------------
#define SW_STRIDE 264   // halves per LDS row (528 B = 33 * 16 B)

__global__ __launch_bounds__(256)
void fused_gemm_tanh(const float* __restrict__ x,
                     const float* __restrict__ mask,
                     const _Float16* __restrict__ Wh,
                     float* __restrict__ out) {
    __shared__ _Float16 sW[256 * SW_STRIDE];   // 132 KB of the 320 KB WGP LDS

    // ---- Stage W' (128 KB) global -> LDS, 16 B per thread per step ----
    {
        const uint4* __restrict__ src = (const uint4*)Wh;   // 8192 chunks
#pragma unroll
        for (int i = 0; i < 32; ++i) {
            const int c    = i * 256 + threadIdx.x;         // coalesced
            const int rowc = c >> 5;                        // W' row
            const int cir  = c & 31;                        // 16B chunk in row
            *(uint4*)(&sW[rowc * SW_STRIDE + cir * 8]) = src[c];
        }
    }
    __syncthreads();

    const int lane   = threadIdx.x & 31;
    const int waveid = threadIdx.x >> 5;
    const int m      = lane & 15;      // M row (A) / N col (B,C,D) within tile
    const int halfid = lane >> 4;

    const int m_base = (blockIdx.x * 8 + waveid) * 16;
    const int row    = m_base + m;

    const float* __restrict__ xrow = x    + ((size_t)row << 8);
    const float* __restrict__ mrow = mask + ((size_t)row << 8);

    // ---- Build the 8 A fragments (A = x - quant(mask), cast to f16) ----
    // ISA 7.12.2 16-bit A 16x32: lane = M, halves 0..7 = K[8*half+0..7],
    // halves 8..15 = K[8*half+16..23], all offset by 32*s.
    v16h af[8];
#pragma unroll
    for (int s = 0; s < 8; ++s) {
        const int k0 = 32 * s + 8 * halfid;
#pragma unroll
        for (int j = 0; j < 8; ++j) {
            af[s][j]     = (_Float16)(xrow[k0 + j]      - dq(mrow[k0 + j]));
            af[s][8 + j] = (_Float16)(xrow[k0 + 16 + j] - dq(mrow[k0 + 16 + j]));
        }
    }

    // ---- 16 output tiles of 16x16 along N, B-fragments from LDS ----
#pragma unroll 1
    for (int nt = 0; nt < 16; ++nt) {
        const _Float16* wr = &sW[(nt * 16 + m) * SW_STRIDE];
        v8f acc = {};
#pragma unroll
        for (int s = 0; s < 8; ++s) {
            const int k0 = 32 * s + 8 * halfid;
            v16h bf;
#pragma unroll
            for (int j = 0; j < 8; ++j) {
                bf[j]     = wr[k0 + j];          // ds_load_b128 (16B aligned)
                bf[8 + j] = wr[k0 + 16 + j];     // ds_load_b128
            }
            acc = __builtin_amdgcn_wmma_f32_16x16x32_f16(
                /*neg_a=*/false, af[s], /*neg_b=*/false, bf,
                /*c_mod=*/(short)0, acc, /*reuse_a=*/false, /*reuse_b=*/false);
        }

        // ---- Epilogue: native tanh + out_mask. C/D: M = r + 8*half, N = m ----
        const int ocol = nt * 16 + m;
#pragma unroll
        for (int r = 0; r < 8; ++r) {
            const int    orow = m_base + 8 * halfid + r;
            const size_t oidx = ((size_t)orow << 8) + ocol;
            out[oidx] = fast_tanh(acc[r]) + dq(mask[oidx]);
        }
    }
}

// ---------------------------------------------------------------------------
extern "C" void kernel_launch(void* const* d_in, const int* in_sizes, int n_in,
                              void* d_out, int out_size, void* d_ws, size_t ws_size,
                              hipStream_t stream) {
    const float* x      = (const float*)d_in[0];
    const float* weight = (const float*)d_in[1];
    const float* mask   = (const float*)d_in[2];
    float*       out    = (float*)d_out;
    _Float16*    Wh     = (_Float16*)d_ws;          // 256*256*2 = 128 KB scratch

    const int IN = 256;
    const int B  = in_sizes[0] / IN;                // 131072

    build_w_f16<<<(256 * 256) / 256, 256, 0, stream>>>(weight, mask, Wh);
    fused_gemm_tanh<<<B / 128, 256, 0, stream>>>(x, mask, Wh, out);
}
